// MetaVisionViT_71012989272469
// MI455X (gfx1250) — matrix operand order repaired and checked by reading, not compiled, not run on hardware
//
#include <hip/hip_runtime.h>
#include <hip/hip_bf16.h>
#include <math.h>

// ---------------------------------------------------------------------------
// CDNA5 (gfx1250) wave32 WMMA types & helpers
// ---------------------------------------------------------------------------
typedef __attribute__((ext_vector_type(16))) __bf16 v16bf;
typedef __attribute__((ext_vector_type(8)))  float  v8f;

__device__ __forceinline__ v8f wmma_bf16(v16bf a, v16bf b, v8f c) {
  // D = A(16x32 bf16) * B(32x16 bf16) + C(16x16 f32)
  return __builtin_amdgcn_wmma_f32_16x16x32_bf16(false, a, false, b, (short)0, c,
                                                 false, false);
}

// A-fragment K index for vgpr-pair j (contiguous: k = 16*hi + 2j, matches B)
__device__ __forceinline__ int afrag_k(int j, int hi) {
  return hi * 16 + (j << 1);
}
// B-fragment K index (contiguous: k = 16*hi + 2j, same order as A)
__device__ __forceinline__ int bfrag_k(int j, int hi) {
  return hi * 16 + (j << 1);
}

// A fragment from row-major fp32 matrix [M x lda], rows row0..row0+15.
// Caller guarantees kb+31 < K (all K are multiples of 32 here).
__device__ __forceinline__ v16bf load_a_rm(const float* __restrict__ A, int lda,
                                           int row0, int kb, int M) {
  const int lane = threadIdx.x & 31, hi = lane >> 4;
  int m = row0 + (lane & 15);
  if (m >= M) m = M - 1;
  const float* Ar = A + (size_t)m * lda + kb;
  v16bf a;
#pragma unroll
  for (int j = 0; j < 8; ++j) {
    int k = afrag_k(j, hi);
    a[2 * j] = (__bf16)Ar[k];
    a[2 * j + 1] = (__bf16)Ar[k + 1];
  }
  return a;
}

// B fragment from bf16 weight stored [N x K] row-major: one contiguous
// 32-byte per-lane vector load (lowered to 2x global_load_b128).
__device__ __forceinline__ v16bf load_b_nk_h(const __bf16* __restrict__ W,
                                             int ldk, int col0, int kb, int N) {
  const int lane = threadIdx.x & 31, hi = lane >> 4;
  int n = col0 + (lane & 15);
  if (n >= N) n = N - 1;
  return *(const v16bf*)(W + (size_t)n * ldk + kb + hi * 16);
}

// B fragment from fp32 matrix stored [N x K] row-major (attention K^T).
// Caller guarantees kb+31 < K.
__device__ __forceinline__ v16bf load_b_nk_f(const float* __restrict__ W,
                                             int ldk, int col0, int kb, int N) {
  const int lane = threadIdx.x & 31, hi = lane >> 4;
  int n = col0 + (lane & 15);
  if (n >= N) n = N - 1;
  const float* Wn = W + (size_t)n * ldk + kb + hi * 16;
  v16bf b;
#pragma unroll
  for (int j = 0; j < 8; ++j) {
    b[2 * j] = (__bf16)Wn[2 * j];
    b[2 * j + 1] = (__bf16)Wn[2 * j + 1];
  }
  return b;
}

// B fragment from fp32 matrix stored [K x N] row-major (attention V), with
// K-bounds masking (zero fill) for the ragged last key block.
__device__ __forceinline__ v16bf load_b_kn_f(const float* __restrict__ W,
                                             int ldn, int col0, int kb, int N,
                                             int K) {
  const int lane = threadIdx.x & 31, hi = lane >> 4;
  int n = col0 + (lane & 15);
  if (n >= N) n = N - 1;
  v16bf b;
#pragma unroll
  for (int j = 0; j < 8; ++j) {
    int k = kb + bfrag_k(j, hi);
    float f0 = (k < K) ? W[(size_t)k * ldn + n] : 0.f;
    float f1 = (k + 1 < K) ? W[(size_t)(k + 1) * ldn + n] : 0.f;
    b[2 * j] = (__bf16)f0;
    b[2 * j + 1] = (__bf16)f1;
  }
  return b;
}

__device__ __forceinline__ float gelu_exact(float x) {
  return 0.5f * x * (1.f + erff(x * 0.70710678118654752f));
}

// ---------------------------------------------------------------------------
// Weight preconversion kernels (run once per launch; ~6us of HBM time).
// ---------------------------------------------------------------------------
// in: L stacked fp32 [K,N] matrices -> out: L stacked bf16 [N,K] matrices
__global__ void k_w_kn2nk(const float* __restrict__ in, __bf16* __restrict__ out,
                          int L, int K, int N) {
  const size_t per = (size_t)K * N;
  const size_t total = per * L;
  for (size_t i = (size_t)blockIdx.x * blockDim.x + threadIdx.x; i < total;
       i += (size_t)gridDim.x * blockDim.x) {
    int l = (int)(i / per);
    size_t rem = i - (size_t)l * per;
    int n = (int)(rem / K);
    int k = (int)(rem - (size_t)n * K);
    out[i] = (__bf16)in[(size_t)l * per + (size_t)k * N + n];
  }
}

// plain fp32 -> bf16 convert (layout preserved)
__global__ void k_w_cvt(const float* __restrict__ in, __bf16* __restrict__ out,
                        size_t total) {
  for (size_t i = (size_t)blockIdx.x * blockDim.x + threadIdx.x; i < total;
       i += (size_t)gridDim.x * blockDim.x)
    out[i] = (__bf16)in[i];
}

// ---------------------------------------------------------------------------
// Generic WMMA GEMM: C[MxN] = act(A[MxK] @ W + bias) (+resid)
// W is preconverted bf16 [N,K].  ACT: 0 none, 1 exact GELU.
// STORE: 0 row-major [M,N], 1 NCHW scatter (M = B*HW, N = channels).
// One wave computes a 32x32 output tile.
// ---------------------------------------------------------------------------
template <int ACT, int STORE>
__global__ void k_gemm(const float* __restrict__ A, const __bf16* __restrict__ W,
                       const float* __restrict__ bias,
                       const float* __restrict__ resid, float* __restrict__ C,
                       int M, int K, int N, int HW) {
  const int lane = threadIdx.x & 31, hi = lane >> 4;
  const int tilesN = (N + 31) / 32;
  const long tiles = (long)((M + 31) / 32) * tilesN;
  long wid = (long)blockIdx.x * (blockDim.x >> 5) + (threadIdx.x >> 5);
  if (wid >= tiles) return;
  const int tm = (int)(wid / tilesN), tn = (int)(wid % tilesN);
  const int row0 = tm * 32, col0 = tn * 32;

  v8f acc00{}, acc01{}, acc10{}, acc11{};
  for (int kb = 0; kb < K; kb += 32) {
    v16bf a0 = load_a_rm(A, K, row0, kb, M);
    v16bf a1 = load_a_rm(A, K, row0 + 16, kb, M);
    v16bf b0 = load_b_nk_h(W, K, col0, kb, N);
    v16bf b1 = load_b_nk_h(W, K, col0 + 16, kb, N);
    if (kb + 32 < K)  // prefetch next A K-slab (global_prefetch_b8)
      __builtin_prefetch(A + (size_t)(row0 + (threadIdx.x & 15)) * K + kb + 32,
                         0, 1);
    acc00 = wmma_bf16(a0, b0, acc00);
    acc01 = wmma_bf16(a0, b1, acc01);
    acc10 = wmma_bf16(a1, b0, acc10);
    acc11 = wmma_bf16(a1, b1, acc11);
  }

#pragma unroll
  for (int mi = 0; mi < 2; ++mi) {
#pragma unroll
    for (int ni = 0; ni < 2; ++ni) {
      v8f acc = (mi == 0) ? (ni == 0 ? acc00 : acc01)
                          : (ni == 0 ? acc10 : acc11);
      int n = col0 + ni * 16 + (lane & 15);
      if (n >= N) continue;
      float bv = bias ? bias[n] : 0.f;
#pragma unroll
      for (int v = 0; v < 8; ++v) {
        int m = row0 + mi * 16 + v + hi * 8;
        if (m >= M) continue;
        float val = acc[v] + bv;
        if (ACT == 1) val = gelu_exact(val);
        if (STORE == 0) {
          size_t idx = (size_t)m * N + n;
          if (resid) val += resid[idx];
          C[idx] = val;
        } else {
          int bb = m / HW, hw = m - bb * HW;
          C[((size_t)bb * N + n) * HW + hw] = val;
        }
      }
    }
  }
}

// ---------------------------------------------------------------------------
// Patch embedding: implicit im2col GEMM.
// M = B*225 tokens, K = 4*8*8 = 256, N = 384. Fuses bias + pos_embed.
// ---------------------------------------------------------------------------
__device__ __forceinline__ v16bf load_a_patch(const float* __restrict__ x,
                                              int row0, int kb) {
  const int lane = threadIdx.x & 31, hi = lane >> 4;
  int m = row0 + (lane & 15);  // M = 14400 is a multiple of 32
  int b = m / 225, p = m - b * 225;
  int ph = p / 15, pw = p - ph * 15;
  const float* xb =
      x + (size_t)b * 4 * 120 * 120 + (size_t)(ph * 8) * 120 + pw * 8;
  v16bf a;
#pragma unroll
  for (int j = 0; j < 8; ++j) {
    int k = kb + afrag_k(j, hi);
#pragma unroll
    for (int e = 0; e < 2; ++e) {
      int kk = k + e;
      int c = kk >> 6, rr = (kk >> 3) & 7, cc = kk & 7;
      a[2 * j + e] = (__bf16)xb[((size_t)c * 120 + rr) * 120 + cc];
    }
  }
  return a;
}

__global__ void k_patch_embed(const float* __restrict__ x,
                              const __bf16* __restrict__ w,  // bf16 [384,256]
                              const float* __restrict__ bias,
                              const float* __restrict__ pos,
                              float* __restrict__ out) {
  const int M = 14400, K = 256, N = 384;
  const int lane = threadIdx.x & 31, hi = lane >> 4;
  const int tilesN = N / 32;  // 12
  int wid = blockIdx.x * (blockDim.x >> 5) + (threadIdx.x >> 5);
  if (wid >= (M / 32) * tilesN) return;
  int tm = wid / tilesN, tn = wid % tilesN;
  int row0 = tm * 32, col0 = tn * 32;

  v8f acc00{}, acc01{}, acc10{}, acc11{};
  for (int kb = 0; kb < K; kb += 32) {
    v16bf a0 = load_a_patch(x, row0, kb);
    v16bf a1 = load_a_patch(x, row0 + 16, kb);
    v16bf b0 = load_b_nk_h(w, K, col0, kb, N);
    v16bf b1 = load_b_nk_h(w, K, col0 + 16, kb, N);
    acc00 = wmma_bf16(a0, b0, acc00);
    acc01 = wmma_bf16(a0, b1, acc01);
    acc10 = wmma_bf16(a1, b0, acc10);
    acc11 = wmma_bf16(a1, b1, acc11);
  }

#pragma unroll
  for (int mi = 0; mi < 2; ++mi) {
#pragma unroll
    for (int ni = 0; ni < 2; ++ni) {
      v8f acc = (mi == 0) ? (ni == 0 ? acc00 : acc01)
                          : (ni == 0 ? acc10 : acc11);
      int n = col0 + ni * 16 + (lane & 15);
      float bv = bias[n];
#pragma unroll
      for (int v = 0; v < 8; ++v) {
        int m = row0 + mi * 16 + v + hi * 8;
        int p = m % 225;
        out[(size_t)m * 384 + n] = acc[v] + bv + pos[(size_t)p * 384 + n];
      }
    }
  }
}

// ---------------------------------------------------------------------------
// LayerNorm: one wave per row of 384 elements.
// ---------------------------------------------------------------------------
__global__ void k_layernorm(const float* __restrict__ x,
                            const float* __restrict__ g,
                            const float* __restrict__ b, float* __restrict__ y,
                            int rows) {
  const int lane = threadIdx.x & 31;
  int r = blockIdx.x * (blockDim.x >> 5) + (threadIdx.x >> 5);
  if (r >= rows) return;
  const float* xr = x + (size_t)r * 384;
  float vals[12];
  float s = 0.f;
#pragma unroll
  for (int i = 0; i < 12; ++i) {
    vals[i] = xr[lane + i * 32];
    s += vals[i];
  }
#pragma unroll
  for (int off = 1; off < 32; off <<= 1) s += __shfl_xor(s, off, 32);
  float mean = s * (1.f / 384.f);
  float s2 = 0.f;
#pragma unroll
  for (int i = 0; i < 12; ++i) {
    float d = vals[i] - mean;
    s2 += d * d;
  }
#pragma unroll
  for (int off = 1; off < 32; off <<= 1) s2 += __shfl_xor(s2, off, 32);
  float inv = rsqrtf(s2 * (1.f / 384.f) + 1e-5f);
  float* yr = y + (size_t)r * 384;
#pragma unroll
  for (int i = 0; i < 12; ++i) {
    int col = lane + i * 32;
    yr[col] = (vals[i] - mean) * inv * g[col] + b[col];
  }
}

// ---------------------------------------------------------------------------
// Flash-style attention. One wave handles (b, h, 16-query tile), streams over
// key blocks of 32 with online softmax; QK^T and P*V both use WMMA. P goes
// through a per-wave LDS tile to re-layout C-frag -> A-frag.
// ---------------------------------------------------------------------------
#define NTOK 225
#define QKV_LD 1152

__device__ __forceinline__ v16bf load_a_ldsP(const __bf16* __restrict__ P) {
  const int lane = threadIdx.x & 31, hi = lane >> 4;
  const int m = lane & 15;
  v16bf a;
#pragma unroll
  for (int j = 0; j < 8; ++j) {
    int k = afrag_k(j, hi);
    a[2 * j] = P[m * 32 + k];
    a[2 * j + 1] = P[m * 32 + k + 1];
  }
  return a;
}

__global__ void k_attention(const float* __restrict__ qkv,
                            float* __restrict__ out) {
  const int lane = threadIdx.x & 31, hi = lane >> 4;
  const int warp = threadIdx.x >> 5;
  int wid = blockIdx.x * 4 + warp;  // 64*6*15 = 5760 waves exactly
  if (wid >= 64 * 6 * 15) return;
  const int qt = wid % 15;
  wid /= 15;
  const int h = wid % 6, b = wid / 6;
  const int q0 = qt * 16;

  __shared__ __bf16 ldsP[4][16 * 32];
  __bf16* P = ldsP[warp];

  const float* Qb = qkv + (size_t)b * NTOK * QKV_LD + h * 64;
  const float* Kb = Qb + 384;
  const float* Vb = Qb + 768;

  // Q fragments, kept in registers for the whole key loop (d = 0..63)
  v16bf aq0 = load_a_rm(Qb, QKV_LD, q0, 0, NTOK);
  v16bf aq1 = load_a_rm(Qb, QKV_LD, q0, 32, NTOK);

  v8f o0{}, o1{}, o2{}, o3{};
  float mrow[8], lrow[8];
#pragma unroll
  for (int v = 0; v < 8; ++v) {
    mrow[v] = -3.0e38f;
    lrow[v] = 0.f;
  }

  for (int kb = 0; kb < NTOK; kb += 32) {
    // scores for this 32-key block (two 16-col tiles), K-dim = head dim 64
    v8f s0{}, s1{};
    {
      v16bf bk = load_b_nk_f(Kb, QKV_LD, kb, 0, NTOK);
      s0 = wmma_bf16(aq0, bk, s0);
      bk = load_b_nk_f(Kb, QKV_LD, kb, 32, NTOK);
      s0 = wmma_bf16(aq1, bk, s0);
      bk = load_b_nk_f(Kb, QKV_LD, kb + 16, 0, NTOK);
      s1 = wmma_bf16(aq0, bk, s1);
      bk = load_b_nk_f(Kb, QKV_LD, kb + 16, 32, NTOK);
      s1 = wmma_bf16(aq1, bk, s1);
    }
    const bool ok0 = (kb + (lane & 15)) < NTOK;
    const bool ok1 = (kb + 16 + (lane & 15)) < NTOK;
#pragma unroll
    for (int v = 0; v < 8; ++v) {
      float x0 = ok0 ? s0[v] * 0.125f : -3.0e38f;  // scale = 64^-0.5
      float x1 = ok1 ? s1[v] * 0.125f : -3.0e38f;
      float t = fmaxf(x0, x1);
#pragma unroll
      for (int off = 1; off < 16; off <<= 1) t = fmaxf(t, __shfl_xor(t, off, 32));
      float mnew = fmaxf(mrow[v], t);
      float corr = __expf(mrow[v] - mnew);
      mrow[v] = mnew;
      float p0 = __expf(x0 - mnew);
      float p1 = __expf(x1 - mnew);
      float rs = p0 + p1;
#pragma unroll
      for (int off = 1; off < 16; off <<= 1) rs += __shfl_xor(rs, off, 32);
      lrow[v] = lrow[v] * corr + rs;
      o0[v] *= corr;
      o1[v] *= corr;
      o2[v] *= corr;
      o3[v] *= corr;
      const int row = v + 8 * hi;
      P[row * 32 + (lane & 15)] = (__bf16)p0;
      P[row * 32 + 16 + (lane & 15)] = (__bf16)p1;
    }
    // O += P(16x32) * V(32x64); V in [key, d] layout = KN for the B fragment
    v16bf pa = load_a_ldsP(P);
    const float* Vblk = Vb + (size_t)kb * QKV_LD;
    const int krem = NTOK - kb;
    v16bf bv = load_b_kn_f(Vblk, QKV_LD, 0, 0, 64, krem);
    o0 = wmma_bf16(pa, bv, o0);
    bv = load_b_kn_f(Vblk, QKV_LD, 16, 0, 64, krem);
    o1 = wmma_bf16(pa, bv, o1);
    bv = load_b_kn_f(Vblk, QKV_LD, 32, 0, 64, krem);
    o2 = wmma_bf16(pa, bv, o2);
    bv = load_b_kn_f(Vblk, QKV_LD, 48, 0, 64, krem);
    o3 = wmma_bf16(pa, bv, o3);
  }

  const int n = lane & 15;
#pragma unroll
  for (int v = 0; v < 8; ++v) {
    int m = q0 + v + 8 * hi;
    if (m < NTOK) {
      float inv = 1.f / lrow[v];
      size_t base = ((size_t)b * NTOK + m) * 384 + h * 64;
      out[base + n] = o0[v] * inv;
      out[base + 16 + n] = o1[v] * inv;
      out[base + 32 + n] = o2[v] * inv;
      out[base + 48 + n] = o3[v] * inv;
    }
  }
}

// ---------------------------------------------------------------------------
// 3x3 SAME conv as implicit GEMM. M = B*H*W, K = Ci*9 (multiple of 32 for all
// decoder stages), N = Co. Zero padding via predicated gathers.
// ---------------------------------------------------------------------------
__global__ void k_conv3(const float* __restrict__ in,
                        const __bf16* __restrict__ w,  // bf16 [Co, Ci*9]
                        const float* __restrict__ bias, float* __restrict__ out,
                        int Bc, int Ci, int Co, int H, int W) {
  const int lane = threadIdx.x & 31, hi = lane >> 4;
  const int HW = H * W;
  const int M = Bc * HW, K = Ci * 9;
  const int tilesN = (Co + 31) / 32;
  const long tiles = (long)((M + 31) / 32) * tilesN;
  long wid = (long)blockIdx.x * (blockDim.x >> 5) + (threadIdx.x >> 5);
  if (wid >= tiles) return;
  const int tm = (int)(wid / tilesN), tn = (int)(wid % tilesN);
  const int row0 = tm * 32, col0 = tn * 32;

  int m0 = row0 + (lane & 15);
  if (m0 >= M) m0 = M - 1;
  int m1 = m0 + 16;
  if (m1 >= M) m1 = M - 1;
  const int b0 = m0 / HW, r0 = m0 - b0 * HW, y0 = r0 / W, x0 = r0 - y0 * W;
  const int b1 = m1 / HW, r1 = m1 - b1 * HW, y1 = r1 / W, x1 = r1 - y1 * W;
  const float* p0 = in + (size_t)b0 * Ci * HW;
  const float* p1 = in + (size_t)b1 * Ci * HW;

  v8f acc00{}, acc01{}, acc10{}, acc11{};
  for (int kb = 0; kb < K; kb += 32) {
    v16bf a0, a1;
#pragma unroll
    for (int j = 0; j < 8; ++j) {
      int k = kb + afrag_k(j, hi);
#pragma unroll
      for (int e = 0; e < 2; ++e) {
        int kk = k + e;
        int ci = kk / 9;
        int rr = kk - ci * 9;
        int q3 = rr / 3;
        int dy = q3 - 1, dx = rr - q3 * 3 - 1;
        int ya = y0 + dy, xa = x0 + dx;
        float fa = (ya >= 0 && ya < H && xa >= 0 && xa < W)
                       ? p0[((size_t)ci * H + ya) * W + xa]
                       : 0.f;
        a0[2 * j + e] = (__bf16)fa;
        int yb = y1 + dy, xb = x1 + dx;
        float fb = (yb >= 0 && yb < H && xb >= 0 && xb < W)
                       ? p1[((size_t)ci * H + yb) * W + xb]
                       : 0.f;
        a1[2 * j + e] = (__bf16)fb;
      }
    }
    v16bf bw0 = load_b_nk_h(w, K, col0, kb, Co);
    v16bf bw1 = load_b_nk_h(w, K, col0 + 16, kb, Co);
    acc00 = wmma_bf16(a0, bw0, acc00);
    acc01 = wmma_bf16(a0, bw1, acc01);
    acc10 = wmma_bf16(a1, bw0, acc10);
    acc11 = wmma_bf16(a1, bw1, acc11);
  }

#pragma unroll
  for (int mi = 0; mi < 2; ++mi) {
#pragma unroll
    for (int ni = 0; ni < 2; ++ni) {
      v8f acc = (mi == 0) ? (ni == 0 ? acc00 : acc01)
                          : (ni == 0 ? acc10 : acc11);
      int n = col0 + ni * 16 + (lane & 15);
      if (n >= Co) continue;
      float bv = bias[n];
#pragma unroll
      for (int v = 0; v < 8; ++v) {
        int m = row0 + mi * 16 + v + hi * 8;
        if (m >= M) continue;
        int bb = m / HW, hw = m - bb * HW;
        out[((size_t)bb * Co + n) * HW + hw] = acc[v] + bv;
      }
    }
  }
}

// ---------------------------------------------------------------------------
// Final 1x1 conv: out[b,o,hw] = sum_c w[o,c]*f[b,c,hw] + b[o]. K=32, N=6.
// One wave = 32 rows x 16 cols (cols >= 6 masked at store).
// ---------------------------------------------------------------------------
__global__ void k_conv1x1(const float* __restrict__ f,
                          const __bf16* __restrict__ w,  // bf16 [6, 32]
                          const float* __restrict__ bias, float* __restrict__ out,
                          int Bc, int Ci, int Co, int HW) {
  const int lane = threadIdx.x & 31, hi = lane >> 4;
  const int M = Bc * HW;
  const int tilesM = (M + 31) / 32;
  int wid = blockIdx.x * (blockDim.x >> 5) + (threadIdx.x >> 5);
  if (wid >= tilesM) return;
  const int row0 = wid * 32;

  v16bf a0, a1;
#pragma unroll
  for (int mi = 0; mi < 2; ++mi) {
    int m = row0 + mi * 16 + (lane & 15);
    if (m >= M) m = M - 1;
    int bb = m / HW, hw = m - bb * HW;
    const float* p = f + (size_t)bb * Ci * HW + hw;
#pragma unroll
    for (int j = 0; j < 8; ++j) {
      int k = afrag_k(j, hi);
      float f0 = p[(size_t)k * HW];
      float f1 = p[(size_t)(k + 1) * HW];
      if (mi == 0) {
        a0[2 * j] = (__bf16)f0;
        a0[2 * j + 1] = (__bf16)f1;
      } else {
        a1[2 * j] = (__bf16)f0;
        a1[2 * j + 1] = (__bf16)f1;
      }
    }
  }
  v16bf bw = load_b_nk_h(w, Ci, 0, 0, Co);
  v8f acc0{}, acc1{};
  acc0 = wmma_bf16(a0, bw, acc0);
  acc1 = wmma_bf16(a1, bw, acc1);

  const int n = lane & 15;
  if (n < Co) {
    float bv = bias[n];
#pragma unroll
    for (int mi = 0; mi < 2; ++mi) {
      v8f acc = mi == 0 ? acc0 : acc1;
#pragma unroll
      for (int v = 0; v < 8; ++v) {
        int m = row0 + mi * 16 + v + hi * 8;
        if (m < M) {
          int bb = m / HW, hw = m - bb * HW;
          out[((size_t)bb * Co + n) * HW + hw] = acc[v] + bv;
        }
      }
    }
  }
}

// ---------------------------------------------------------------------------
// Bilinear 2x upsample (half-pixel centers, align_corners=False, edge clamp)
// ---------------------------------------------------------------------------
__global__ void k_upsample2x(const float* __restrict__ in, float* __restrict__ out,
                             int BC, int H, int W) {
  const int H2 = H * 2, W2 = W * 2;
  const size_t total = (size_t)BC * H2 * W2;
  for (size_t i = (size_t)blockIdx.x * blockDim.x + threadIdx.x; i < total;
       i += (size_t)gridDim.x * blockDim.x) {
    int bc = (int)(i / ((size_t)H2 * W2));
    int rem = (int)(i - (size_t)bc * H2 * W2);
    int y = rem / W2, x = rem - y * W2;
    float sy = (y + 0.5f) * 0.5f - 0.5f;
    float sx = (x + 0.5f) * 0.5f - 0.5f;
    int yf = (int)floorf(sy), xf = (int)floorf(sx);
    float wy = sy - yf, wx = sx - xf;
    int ya = min(max(yf, 0), H - 1), yb = min(max(yf + 1, 0), H - 1);
    int xa = min(max(xf, 0), W - 1), xb = min(max(xf + 1, 0), W - 1);
    const float* p = in + (size_t)bc * H * W;
    float v00 = p[(size_t)ya * W + xa], v01 = p[(size_t)ya * W + xb];
    float v10 = p[(size_t)yb * W + xa], v11 = p[(size_t)yb * W + xb];
    out[i] = (1.f - wy) * ((1.f - wx) * v00 + wx * v01) +
             wy * ((1.f - wx) * v10 + wx * v11);
  }
}

// ---------------------------------------------------------------------------
// Training-mode BatchNorm: per-channel stats then normalize + ReLU (in place)
// ---------------------------------------------------------------------------
__global__ void k_bn_stats(const float* __restrict__ x, float* __restrict__ stats,
                           int Bc, int C, int HW) {
  const int c = blockIdx.x;
  const size_t n = (size_t)Bc * HW;
  float s = 0.f, s2 = 0.f;
  for (size_t i = threadIdx.x; i < n; i += blockDim.x) {
    int b = (int)(i / HW);
    int hw = (int)(i - (size_t)b * HW);
    float v = x[((size_t)b * C + c) * HW + hw];
    s += v;
    s2 += v * v;
  }
  __shared__ float sh0[256], sh1[256];
  sh0[threadIdx.x] = s;
  sh1[threadIdx.x] = s2;
  __syncthreads();
  for (int off = 128; off > 0; off >>= 1) {
    if ((int)threadIdx.x < off) {
      sh0[threadIdx.x] += sh0[threadIdx.x + off];
      sh1[threadIdx.x] += sh1[threadIdx.x + off];
    }
    __syncthreads();
  }
  if (threadIdx.x == 0) {
    float mean = sh0[0] / (float)n;
    float var = sh1[0] / (float)n - mean * mean;
    stats[c] = mean;
    stats[C + c] = rsqrtf(var + 1e-5f);
  }
}

__global__ void k_bn_apply(float* __restrict__ x, const float* __restrict__ stats,
                           const float* __restrict__ g, const float* __restrict__ b,
                           int C, int HW, size_t total) {
  for (size_t i = (size_t)blockIdx.x * blockDim.x + threadIdx.x; i < total;
       i += (size_t)gridDim.x * blockDim.x) {
    int c = (int)((i / HW) % C);
    float v = (x[i] - stats[c]) * stats[C + c] * g[c] + b[c];
    x[i] = fmaxf(v, 0.f);
  }
}

// ---------------------------------------------------------------------------
// Host side
// ---------------------------------------------------------------------------
extern "C" void kernel_launch(void* const* d_in, const int* in_sizes, int n_in,
                              void* d_out, int out_size, void* d_ws,
                              size_t ws_size, hipStream_t stream) {
  (void)in_sizes; (void)n_in; (void)out_size; (void)ws_size;

  const float* x        = (const float*)d_in[0];
  const float* patch_w  = (const float*)d_in[1];
  const float* patch_b  = (const float*)d_in[2];
  const float* pos      = (const float*)d_in[3];
  const float* ln1_g    = (const float*)d_in[4];
  const float* ln1_b    = (const float*)d_in[5];
  const float* qkv_w    = (const float*)d_in[6];
  const float* qkv_b    = (const float*)d_in[7];
  const float* proj_w   = (const float*)d_in[8];
  const float* proj_b   = (const float*)d_in[9];
  const float* ln2_g    = (const float*)d_in[10];
  const float* ln2_b    = (const float*)d_in[11];
  const float* fc1_w    = (const float*)d_in[12];
  const float* fc1_b    = (const float*)d_in[13];
  const float* fc2_w    = (const float*)d_in[14];
  const float* fc2_b    = (const float*)d_in[15];
  const float* enc_g    = (const float*)d_in[16];
  const float* enc_b    = (const float*)d_in[17];
  const float* dproj_w  = (const float*)d_in[18];
  const float* dproj_b  = (const float*)d_in[19];
  const float* out_w    = (const float*)d_in[44];
  const float* out_b    = (const float*)d_in[45];

  const int Mtok = 14400;  // B * 225
  constexpr size_t SZ_TOK = 14400ull * 384;
  constexpr size_t SZ_HID = 14400ull * 1536;
  constexpr size_t SZ_DEC = 64ull * 64 * 120 * 120;

  float* ws   = (float*)d_ws;
  float* PING = ws;
  float* PONG = ws + SZ_DEC;
  float* T0   = ws + 2 * SZ_DEC;           // tokens
  float* T1   = T0 + SZ_TOK;               // scratch (ln out / new tokens)
  float* T2   = T1 + SZ_TOK;               // qkv / mlp hidden
  float* T3   = T2 + SZ_HID;               // attn out / ln2 out
  float* STATS = T3 + SZ_TOK;              // bn stats (2*C floats)

  // bf16 weight arena (32B-aligned: every sub-size is a multiple of 16 elems)
  __bf16* WB = (__bf16*)(STATS + 512);
  size_t wo = 0;
  __bf16* wqkv  = WB + wo; wo += 12ull * 1152 * 384;
  __bf16* wproj = WB + wo; wo += 12ull * 384 * 384;
  __bf16* wfc1  = WB + wo; wo += 12ull * 1536 * 384;
  __bf16* wfc2  = WB + wo; wo += 12ull * 384 * 1536;
  __bf16* wpat  = WB + wo; wo += 384ull * 256;
  __bf16* wdp   = WB + wo; wo += 256ull * 384;
  __bf16* wcv[6];
  const size_t cvsz[6] = {128ull * 256 * 9, 128ull * 128 * 9, 64ull * 128 * 9,
                          64ull * 64 * 9,   32ull * 64 * 9,   32ull * 32 * 9};
  for (int i = 0; i < 6; ++i) { wcv[i] = WB + wo; wo += cvsz[i]; }
  __bf16* wout = WB + wo; wo += 192;

  auto blocksFor = [](long waves) { return (int)((waves + 3) / 4); };

  // --- one-time-per-launch weight conversion to bf16 [N,K]
  k_w_kn2nk<<<1024, 256, 0, stream>>>(qkv_w, wqkv, 12, 384, 1152);
  k_w_kn2nk<<<1024, 256, 0, stream>>>(proj_w, wproj, 12, 384, 384);
  k_w_kn2nk<<<1024, 256, 0, stream>>>(fc1_w, wfc1, 12, 384, 1536);
  k_w_kn2nk<<<1024, 256, 0, stream>>>(fc2_w, wfc2, 12, 1536, 384);
  k_w_cvt<<<96, 256, 0, stream>>>(patch_w, wpat, 384ull * 256);
  k_w_cvt<<<96, 256, 0, stream>>>(dproj_w, wdp, 256ull * 384);
  for (int s = 0; s < 3; ++s) {
    k_w_cvt<<<96, 256, 0, stream>>>((const float*)d_in[20 + s * 8 + 0],
                                    wcv[2 * s], cvsz[2 * s]);
    k_w_cvt<<<96, 256, 0, stream>>>((const float*)d_in[20 + s * 8 + 4],
                                    wcv[2 * s + 1], cvsz[2 * s + 1]);
  }
  k_w_cvt<<<1, 192, 0, stream>>>(out_w, wout, 192);

  // --- patch embed: x -> T0 tokens [14400, 384]
  k_patch_embed<<<blocksFor(450L * 12), 128, 0, stream>>>(x, wpat, patch_b, pos,
                                                          T0);

  // --- 12 transformer blocks
  for (int l = 0; l < 12; ++l) {
    const float* g1 = ln1_g + (size_t)l * 384;
    const float* b1 = ln1_b + (size_t)l * 384;
    const float* qb = qkv_b + (size_t)l * 1152;
    const float* pb = proj_b + (size_t)l * 384;
    const float* g2 = ln2_g + (size_t)l * 384;
    const float* b2 = ln2_b + (size_t)l * 384;
    const float* f1b = fc1_b + (size_t)l * 1536;
    const float* f2b = fc2_b + (size_t)l * 384;
    const __bf16* qw  = wqkv + (size_t)l * 1152 * 384;
    const __bf16* pw  = wproj + (size_t)l * 384 * 384;
    const __bf16* f1w = wfc1 + (size_t)l * 1536 * 384;
    const __bf16* f2w = wfc2 + (size_t)l * 384 * 1536;

    k_layernorm<<<Mtok / 4, 128, 0, stream>>>(T0, g1, b1, T1, Mtok);
    k_gemm<0, 0><<<blocksFor(450L * 36), 128, 0, stream>>>(
        T1, qw, qb, nullptr, T2, Mtok, 384, 1152, 0);
    k_attention<<<1440, 128, 0, stream>>>(T2, T3);
    k_gemm<0, 0><<<blocksFor(450L * 12), 128, 0, stream>>>(
        T3, pw, pb, /*resid=*/T0, T1, Mtok, 384, 384, 0);
    k_layernorm<<<Mtok / 4, 128, 0, stream>>>(T1, g2, b2, T3, Mtok);
    k_gemm<1, 0><<<blocksFor(450L * 48), 128, 0, stream>>>(
        T3, f1w, f1b, nullptr, T2, Mtok, 384, 1536, 0);
    k_gemm<0, 0><<<blocksFor(450L * 12), 128, 0, stream>>>(
        T2, f2w, f2b, /*resid=*/T1, T0, Mtok, 1536, 384, 0);
  }

  // --- final encoder LN, then dec_proj with NCHW scatter: -> PING [64,256,15,15]
  k_layernorm<<<Mtok / 4, 128, 0, stream>>>(T0, enc_g, enc_b, T1, Mtok);
  k_gemm<0, 1><<<blocksFor(450L * 8), 128, 0, stream>>>(
      T1, wdp, dproj_b, nullptr, PING, Mtok, 384, 256, /*HW=*/225);

  // --- decoder: 3 stages of (upsample2x, conv3+bn+relu, conv3+bn+relu)
  float* src = PING;
  float* dst = PONG;
  int Hs = 15;
  const int chans[4] = {256, 128, 64, 32};
  for (int s = 0; s < 3; ++s) {
    const int cin = chans[s], cout = chans[s + 1];
    const int Ho = Hs * 2;
    const float* c1b = (const float*)d_in[20 + s * 8 + 1];
    const float* bg1 = (const float*)d_in[20 + s * 8 + 2];
    const float* bb1 = (const float*)d_in[20 + s * 8 + 3];
    const float* c2b = (const float*)d_in[20 + s * 8 + 5];
    const float* bg2 = (const float*)d_in[20 + s * 8 + 6];
    const float* bb2 = (const float*)d_in[20 + s * 8 + 7];

    k_upsample2x<<<2048, 256, 0, stream>>>(src, dst, 64 * cin, Hs, Hs);

    long tiles1 = (long)((64L * Ho * Ho + 31) / 32) * ((cout + 31) / 32);
    k_conv3<<<blocksFor(tiles1), 128, 0, stream>>>(dst, wcv[2 * s], c1b, src, 64,
                                                   cin, cout, Ho, Ho);
    k_bn_stats<<<cout, 256, 0, stream>>>(src, STATS, 64, cout, Ho * Ho);
    k_bn_apply<<<2048, 256, 0, stream>>>(src, STATS, bg1, bb1, cout, Ho * Ho,
                                         (size_t)64 * cout * Ho * Ho);

    long tiles2 = (long)((64L * Ho * Ho + 31) / 32) * ((cout + 31) / 32);
    k_conv3<<<blocksFor(tiles2), 128, 0, stream>>>(src, wcv[2 * s + 1], c2b, dst,
                                                   64, cout, cout, Ho, Ho);
    k_bn_stats<<<cout, 256, 0, stream>>>(dst, STATS, 64, cout, Ho * Ho);
    k_bn_apply<<<2048, 256, 0, stream>>>(dst, STATS, bg2, bb2, cout, Ho * Ho,
                                         (size_t)64 * cout * Ho * Ho);

    float* t = src; src = dst; dst = t;  // stage output now in src
    Hs = Ho;
  }

  // --- final 1x1 conv: src [64,32,120,120] -> d_out [64,6,120,120]
  k_conv1x1<<<blocksFor(28800L), 128, 0, stream>>>(src, wout, out_b,
                                                   (float*)d_out, 64, 32, 6,
                                                   14400);
}
